// attention_nearly_Layer_90993177133839
// MI455X (gfx1250) — matrix operand order repaired and checked by reading, not compiled, not run on hardware
//
#include <hip/hip_runtime.h>

// ---------------------------------------------------------------------------
// Bahdanau additive attention for MI455X (gfx1250, wave32, WMMA).
//
// Shapes: B=32, T=2048, D=1024, UNITS=1024.
// Dominant cost: values@W1 = 137 GFLOP. values = 256MB fp32 (read 2x -> ~22us
// HBM floor at 23.3 TB/s). bf16 WMMA (16x16x32) gives 8x the per-instruction
// FLOPs of fp32 WMMA (16x16x4), making compute ~match the memory floor.
// h [B,T,U] (256MB) is never materialized: GEMM -> tanh -> dot(V_w) is fused.
// V_b is softmax-invariant and dropped.
// tanh is computed branch-free from native v_exp_f32/v_rcp_f32 so the epilogue
// co-executes with the WMMA (TRANS-tracked) pipe instead of libm's saveexec
// branch ladder.
// ---------------------------------------------------------------------------

#define B_SZ 32
#define T_SZ 2048
#define D_SZ 1024
#define U_SZ 1024

typedef __bf16 bf16;
typedef bf16  v16bf __attribute__((ext_vector_type(16)));
typedef bf16  v8bf  __attribute__((ext_vector_type(8)));
typedef float v8f   __attribute__((ext_vector_type(8)));

__device__ __forceinline__ unsigned short f32_to_bf16_bits(float f) {
    unsigned int u = __builtin_bit_cast(unsigned int, f);
    u += 0x7FFFu + ((u >> 16) & 1u);   // round-to-nearest-even
    return (unsigned short)(u >> 16);
}

// Branch-free tanh: 1 - 2/(e^{2x}+1), via native exp2/rcp.
// Saturates to +/-1 for large |x| (inf->rcp 0; 0->rcp 1), NaN propagates.
__device__ __forceinline__ float fast_tanh(float x) {
    float e = __builtin_amdgcn_exp2f(x * 2.88539008177792681472f);  // 2*log2(e)
    float r = __builtin_amdgcn_rcpf(e + 1.0f);
    return __builtin_fmaf(-2.0f, r, 1.0f);
}

// ---------------- kernel 1: W1 fp32 -> bf16 (2MB workspace) ----------------
__global__ __launch_bounds__(256) void convw1_kernel(const float* __restrict__ w,
                                                     bf16* __restrict__ out) {
    size_t i = (size_t)blockIdx.x * 256 + threadIdx.x;      // D*U/4 float4's
    float4 f = ((const float4*)w)[i];
    unsigned p0 = (unsigned)f32_to_bf16_bits(f.x) | ((unsigned)f32_to_bf16_bits(f.y) << 16);
    unsigned p1 = (unsigned)f32_to_bf16_bits(f.z) | ((unsigned)f32_to_bf16_bits(f.w) << 16);
    ((uint2*)out)[i] = make_uint2(p0, p1);
}

// -------- kernel 2: qbias[b,u] = query[b]@W2 + W1_b + W2_b  (tiny) ---------
__global__ __launch_bounds__(256) void qbias_kernel(const float* __restrict__ q,
                                                    const float* __restrict__ W2,
                                                    const float* __restrict__ b1,
                                                    const float* __restrict__ b2,
                                                    float* __restrict__ qbias) {
    __shared__ float sq[D_SZ];
    int b = blockIdx.y;
    int u = blockIdx.x * 256 + threadIdx.x;
    for (int i = threadIdx.x; i < D_SZ; i += 256) sq[i] = q[(size_t)b * D_SZ + i];
    __syncthreads();
    float acc = b1[u] + b2[u];
#pragma unroll 8
    for (int d = 0; d < D_SZ; ++d) acc += sq[d] * W2[(size_t)d * U_SZ + u];
    qbias[(size_t)b * U_SZ + u] = acc;
}

// ---- kernel 3: fused score = V_w . tanh(values@W1 + qbias)   [WMMA bf16] ----
// Block = 8 waves x 16 T-rows = 128 rows of one batch. Values tile staged once
// in LDS as bf16 (row pad +8 elems -> 16B shift/row -> conflict-free b128
// fragment loads). Per 16-wide u-tile, a transposed W1 slab is staged and each
// wave runs K=1024 as 32x v_wmma_f32_16x16x32_bf16.
#define BT_TILE 128
#define STR     (D_SZ + 8)   // padded LDS row stride in bf16 elements

__global__ __launch_bounds__(256) void score_kernel(const float* __restrict__ values,
                                                    const bf16*  __restrict__ w1bf,
                                                    const float* __restrict__ qbias,
                                                    const float* __restrict__ Vw,
                                                    float* __restrict__ scores) {
    extern __shared__ unsigned char smem[];
    bf16*  ldsV  = (bf16*)smem;                 // [128][STR] values tile (bf16)
    bf16*  ldsW  = ldsV + BT_TILE * STR;        // [16][STR]  W1 slab, transposed [n][k]
    float* ldsVw = (float*)(ldsW + 16 * STR);   // [U] V_w
    float* ldsQb = ldsVw + U_SZ;                // [U] qbias row for this b

    const int tid  = threadIdx.x;
    const int b    = blockIdx.y;
    const int t0   = blockIdx.x * BT_TILE;
    const int lane = tid & 31;
    const int wave = tid >> 5;
    const int n    = lane & 15;   // column within 16-wide tile / A-row id
    const int g    = lane >> 4;   // lane half

    // ---- stage values[b, t0:t0+128, :] -> bf16 LDS (coalesced float4) ----
    const float4* v4 = (const float4*)(values + ((size_t)b * T_SZ + t0) * D_SZ);
    for (int i = tid; i < BT_TILE * (D_SZ / 4); i += 256) {
        int row = i >> 8;            // D/4 = 256
        int c4  = i & 255;
        float4 f = v4[i];
        unsigned p0 = (unsigned)f32_to_bf16_bits(f.x) | ((unsigned)f32_to_bf16_bits(f.y) << 16);
        unsigned p1 = (unsigned)f32_to_bf16_bits(f.z) | ((unsigned)f32_to_bf16_bits(f.w) << 16);
        *(uint2*)(ldsV + (size_t)row * STR + c4 * 4) = make_uint2(p0, p1);
    }
    for (int i = tid; i < U_SZ; i += 256) {
        ldsVw[i] = Vw[i];
        ldsQb[i] = qbias[(size_t)b * U_SZ + i];
    }

    float spart[8] = {0, 0, 0, 0, 0, 0, 0, 0};
    const bf16* aB = ldsV + (size_t)(wave * 16 + n) * STR + 8 * g;  // A row base

    for (int ut = 0; ut < U_SZ / 16; ++ut) {
        __syncthreads();   // also covers initial staging on ut==0
        // stage W1[:, ut*16 .. +16] transposed: ldsW[n][k]
        const bf16* wg = w1bf + ut * 16;
        for (int i = tid; i < 16 * D_SZ; i += 256) {
            int nn = i & 15, k = i >> 4;
            ldsW[(size_t)nn * STR + k] = wg[(size_t)k * U_SZ + nn];
        }
        if (ut + 1 < U_SZ / 16)
            __builtin_prefetch(w1bf + (ut + 1) * 16, 0, 1);  // global_prefetch_b8
        __syncthreads();

        v8f acc = {0, 0, 0, 0, 0, 0, 0, 0};
        const bf16* bB = ldsW + (size_t)n * STR + 8 * g;     // B col base
#pragma unroll 8
        for (int kc = 0; kc < D_SZ / 32; ++kc) {
            // A 16x32 bf16 lane layout: a[0..7]=K(8g+0..7), a[8..15]=K(8g+16..23)
            v8bf alo = *(const v8bf*)(aB + kc * 32);
            v8bf ahi = *(const v8bf*)(aB + kc * 32 + 16);
            // B 32x16 bf16 lane layout: b[0..7]=K(8g+0..7), b[8..15]=K(8g+16..23)
            v8bf blo = *(const v8bf*)(bB + kc * 32);
            v8bf bhi = *(const v8bf*)(bB + kc * 32 + 16);
            v16bf Af, Bf;
#pragma unroll
            for (int e = 0; e < 8; ++e) {
                Af[e] = alo[e]; Af[e + 8] = ahi[e];
                Bf[e] = blo[e]; Bf[e + 8] = bhi[e];
            }
            acc = __builtin_amdgcn_wmma_f32_16x16x32_bf16(
                false, Af, false, Bf, (short)0, acc, false, false);
        }
        // fuse tanh + dot with V_w. C layout: lane(n,g), VGPR r -> M = r + 8g,
        // N = n; column u = ut*16 + n is fixed per lane. Branch-free tanh.
        float qb = ldsQb[ut * 16 + n];
        float vw = ldsVw[ut * 16 + n];
#pragma unroll
        for (int r = 0; r < 8; ++r) spart[r] += fast_tanh(acc[r] + qb) * vw;
    }

    // reduce the 16 N-columns per lane-half (xor masks < 16 stay in-half)
#pragma unroll
    for (int r = 0; r < 8; ++r) {
        float s = spart[r];
        s += __shfl_xor(s, 1);
        s += __shfl_xor(s, 2);
        s += __shfl_xor(s, 4);
        s += __shfl_xor(s, 8);
        spart[r] = s;
    }
    if (n == 0) {  // lanes 0 (rows 0..7) and 16 (rows 8..15)
        float* sp = scores + (size_t)b * T_SZ + t0 + wave * 16 + g * 8;
#pragma unroll
        for (int r = 0; r < 8; ++r) sp[r] = spart[r];
    }
}

// ---------------- kernel 4: softmax over T, in place in d_out ---------------
__global__ __launch_bounds__(256) void softmax_kernel(float* __restrict__ attn) {
    __shared__ float red[256];
    int b = blockIdx.x, tid = threadIdx.x;
    float* s = attn + (size_t)b * T_SZ;
    float x[8], m = -3.402823466e+38f;
#pragma unroll
    for (int k = 0; k < 8; ++k) { x[k] = s[tid + k * 256]; m = fmaxf(m, x[k]); }
    red[tid] = m; __syncthreads();
    for (int off = 128; off > 0; off >>= 1) {
        if (tid < off) red[tid] = fmaxf(red[tid], red[tid + off]);
        __syncthreads();
    }
    m = red[0]; __syncthreads();
    float sum = 0.f;
#pragma unroll
    for (int k = 0; k < 8; ++k) {
        x[k] = __builtin_amdgcn_exp2f((x[k] - m) * 1.44269504088896340736f);
        sum += x[k];
    }
    red[tid] = sum; __syncthreads();
    for (int off = 128; off > 0; off >>= 1) {
        if (tid < off) red[tid] += red[tid + off];
        __syncthreads();
    }
    float inv = 1.0f / red[0];
#pragma unroll
    for (int k = 0; k < 8; ++k) s[tid + k * 256] = x[k] * inv;
}

// ------- kernel 5: context partials over T chunks (no atomics; det.) --------
__global__ __launch_bounds__(256) void ctx_partial_kernel(const float* __restrict__ values,
                                                          const float* __restrict__ attn,
                                                          float* __restrict__ partial) {
    __shared__ float wsh[256];
    int b = blockIdx.z, tc = blockIdx.y, dc = blockIdx.x, tid = threadIdx.x;
    wsh[tid] = attn[(size_t)b * T_SZ + tc * 256 + tid];
    __syncthreads();
    int d = dc * 256 + tid;
    const float* vp = values + ((size_t)b * T_SZ + (size_t)tc * 256) * D_SZ + d;
    float acc = 0.f;
#pragma unroll 8
    for (int tt = 0; tt < 256; ++tt) acc += wsh[tt] * vp[(size_t)tt * D_SZ];
    partial[((size_t)b * (T_SZ / 256) + tc) * D_SZ + d] = acc;
}

__global__ __launch_bounds__(256) void ctx_reduce_kernel(const float* __restrict__ partial,
                                                         float* __restrict__ ctx) {
    int b = blockIdx.y;
    int d = blockIdx.x * 256 + threadIdx.x;
    float acc = 0.f;
#pragma unroll
    for (int tc = 0; tc < T_SZ / 256; ++tc)
        acc += partial[((size_t)b * (T_SZ / 256) + tc) * D_SZ + d];
    ctx[(size_t)b * D_SZ + d] = acc;
}

// ---------------------------------------------------------------------------
extern "C" void kernel_launch(void* const* d_in, const int* in_sizes, int n_in,
                              void* d_out, int out_size, void* d_ws, size_t ws_size,
                              hipStream_t stream) {
    const float* query = (const float*)d_in[0];   // [B,1,D]
    const float* values = (const float*)d_in[1];  // [B,T,D]
    const float* W1w = (const float*)d_in[2];     // [D,U]
    const float* W1b = (const float*)d_in[3];     // [U]
    const float* W2w = (const float*)d_in[4];     // [D,U]
    const float* W2b = (const float*)d_in[5];     // [U]
    const float* Vw  = (const float*)d_in[6];     // [U,1]
    // d_in[7] = V_b : constant shift of all scores -> softmax-invariant, unused.

    float* ctx  = (float*)d_out;                        // [B,D]
    float* attn = (float*)d_out + (size_t)B_SZ * D_SZ;  // [B,T] (scores, then in-place softmax)

    // workspace: W1 bf16 (2MB) | qbias (128KB) | context partials (1MB)
    bf16*  w1bf    = (bf16*)d_ws;
    float* qbias   = (float*)((char*)d_ws + (size_t)D_SZ * U_SZ * sizeof(bf16));
    float* partial = (float*)((char*)qbias + (size_t)B_SZ * U_SZ * sizeof(float));

    const size_t smem_score =
        (size_t)(BT_TILE * STR + 16 * STR) * sizeof(bf16) + 2u * U_SZ * sizeof(float); // ~298KB < 320KB WGP LDS
    hipFuncSetAttribute(reinterpret_cast<const void*>(score_kernel),
                        hipFuncAttributeMaxDynamicSharedMemorySize, (int)smem_score);

    convw1_kernel<<<(D_SZ * U_SZ / 4) / 256, 256, 0, stream>>>(W1w, w1bf);
    qbias_kernel<<<dim3(U_SZ / 256, B_SZ), 256, 0, stream>>>(query, W2w, W1b, W2b, qbias);
    score_kernel<<<dim3(T_SZ / BT_TILE, B_SZ), 256, smem_score, stream>>>(values, w1bf, qbias, Vw, attn);
    softmax_kernel<<<B_SZ, 256, 0, stream>>>(attn);
    ctx_partial_kernel<<<dim3(D_SZ / 256, T_SZ / 256, B_SZ), 256, 0, stream>>>(values, attn, partial);
    ctx_reduce_kernel<<<dim3(D_SZ / 256, B_SZ), 256, 0, stream>>>(partial, ctx);
}
